// ST_BLOCK_5_58832462021176
// MI455X (gfx1250) — matrix operand order repaired and checked by reading, not compiled, not run on hardware
//
#include <hip/hip_runtime.h>
#include <hip/hip_bf16.h>
#include <math.h>

// ------------------------------------------------------------------
// ST-Block graph-conv LSTM for MI455X (gfx1250, wave32, WMMA bf16)
// B=64, C_IN=32, C_OUT=64, N=512, T=24, K_CHEB=3 (k=0 is identity)
// Round 2: LDS-stage the block-shared A-side operands (tem tile in
// k_spatial, W row-blocks in k_gates) -> 8x less redundant global
// traffic, A fragments now come from ds_load_b128.
// ------------------------------------------------------------------

typedef __bf16 bf16_t;
typedef __attribute__((ext_vector_type(16))) __bf16 v16bf;
typedef __attribute__((ext_vector_type(8)))  __bf16 v8bf;
typedef __attribute__((ext_vector_type(8)))  float  v8f;

#define B_DIM 64
#define C_IN  32
#define C_OUT 64
#define N_DIM 512
#define T_DIM 24
#define CC    288   // K_CHEB * (C_IN + C_OUT)

// ---- WMMA fragment loaders (contiguous k order, not the ISA table order; A and B match) ----------
// A (16x32 bf16): lane row = lane&15; lanes<16 hold K 0-15,
// lanes>=16 hold K 16-31 -> two 16B contiguous chunks per lane.
// Works for both global and LDS-resident tiles (addrspace inferred).
template <typename PTR>
static __device__ __forceinline__ v16bf load_a_frag(PTR base, int ld, int lane, int k0) {
  const int row  = lane & 15;
  const int half = lane >> 4;
  const int off  = row * ld + k0 + half * 16;
  v8bf lo, hi;
  lo = *(const v8bf*)(base + off);
  hi = *(const v8bf*)(base + off + 8);
  v16bf r;
#pragma unroll
  for (int i = 0; i < 8; ++i) { r[i] = lo[i]; r[i + 8] = hi[i]; }
  return r;
}

// B (32x16 bf16) from storage laid out [n_out][k] row-major:
// lane column = lane&15; lanes<16 hold K 0-15, lanes>=16 hold K 16-31
// -> one 32B contiguous run per lane.
static __device__ __forceinline__ v16bf load_b_frag(const bf16_t* __restrict__ base,
                                                    int ld, int lane, int k0) {
  const int row  = lane & 15;
  const int half = lane >> 4;
  const bf16_t* p = base + (size_t)row * ld + k0 + half * 16;
  const v8bf lo = *(const v8bf*)(p);
  const v8bf hi = *(const v8bf*)(p + 8);
  v16bf r;
#pragma unroll
  for (int i = 0; i < 8; ++i) { r[i] = lo[i]; r[i + 8] = hi[i]; }
  return r;
}

static __device__ __forceinline__ float sigf(float x) {
  return 1.0f / (1.0f + __expf(-x));
}

// ---- one-time prep kernels ----------------------------------------

// x[B,C,N,T] fp32 -> xbf[T,B,C,N] bf16 (coalesced writes along n)
__global__ void k_pack_x(const float* __restrict__ x, bf16_t* __restrict__ xbf) {
  size_t idx = (size_t)blockIdx.x * blockDim.x + threadIdx.x; // 24*2^20 total
  int n = (int)(idx & 511);
  int c = (int)((idx >> 9) & 31);
  int b = (int)((idx >> 14) & 63);
  int t = (int)(idx >> 20);
  xbf[idx] = (bf16_t)x[(((size_t)b * C_IN + c) * N_DIM + n) * T_DIM + t];
}

__global__ void k_pack_w(const float* __restrict__ W, bf16_t* __restrict__ Wbf) {
  int idx = blockIdx.x * blockDim.x + threadIdx.x; // 256*288
  Wbf[idx] = (bf16_t)W[idx];
}

// Chebyshev operators stored as Ls[k] row-major == [n_out][n_in] layout
// (exactly the B-fragment layout; Lap[k] = Ls[k]^T cancels the transpose).
// L1 = S ; L2 = 2*S@S - I.  One-time 512^3 fp32 GEMM, bandwidth-trivial.
__global__ void k_laps(const float* __restrict__ S,
                       bf16_t* __restrict__ L1, bf16_t* __restrict__ L2) {
  int idx = blockIdx.x * blockDim.x + threadIdx.x; // 512*512
  int p = idx >> 9, q = idx & 511;
  L1[idx] = (bf16_t)S[idx];
  float acc = 0.0f;
  for (int r = 0; r < N_DIM; ++r)
    acc += S[p * N_DIM + r] * S[r * N_DIM + q];
  acc = 2.0f * acc - (p == q ? 1.0f : 0.0f);
  L2[idx] = (bf16_t)acc;
}

__global__ void k_init_state(bf16_t* __restrict__ h, float* __restrict__ c) {
  size_t idx = (size_t)blockIdx.x * blockDim.x + threadIdx.x; // B*C_OUT*N
  h[idx] = (bf16_t)0.0f;
  c[idx] = 0.0f;
}

// ---- per-step kernel 1: spatial graph conv ------------------------
// G[k] = tem(96x512) @ Mk(512x512) for k=1,2 ; k=0 is identity copy.
// All 8 waves of a block share (b, c-tile): the 16x512 tem tile is
// staged in LDS once and A fragments come from LDS; the per-wave
// L1/L2 B fragments stream from global (L2-resident, 1 MB total).
// Output written bf16-transposed into Zt[b][n][cc], cc = c*3 + k.
__global__ void __launch_bounds__(256) k_spatial(
    const bf16_t* __restrict__ xbf, const bf16_t* __restrict__ hbf,
    const bf16_t* __restrict__ L1,  const bf16_t* __restrict__ L2,
    bf16_t* __restrict__ Zt, int t) {
  __shared__ __align__(16) bf16_t stem[16 * N_DIM];   // 16 KB

  const int lane = threadIdx.x & 31;
  const int wave = threadIdx.x >> 5;
  const int bid  = blockIdx.x;        // B * 6 * 4
  const int b     = bid / 24;
  const int rem   = bid % 24;
  const int ctile = rem >> 2;
  const int ng    = rem & 3;
  const int n0 = (ng * 8 + wave) * 16;
  const int c0 = ctile * 16;

  // tem rows: c<32 from x slice, c>=32 from recurrent h (tiles never
  // straddle the boundary); the 16x512 tile is one contiguous block.
  const bf16_t* tembase = (c0 < C_IN)
      ? xbf + ((size_t)t * B_DIM + b) * C_IN * N_DIM + (size_t)c0 * N_DIM
      : hbf + (size_t)b * C_OUT * N_DIM + (size_t)(c0 - C_IN) * N_DIM;

  // cooperative stage: 8192 bf16 = 1024 x 16B chunks, 4 per thread
#pragma unroll
  for (int i = 0; i < 4; ++i) {
    const int off = (i * 256 + (int)threadIdx.x) * 8;
    *(v8bf*)(stem + off) = *(const v8bf*)(tembase + off);
  }
  __syncthreads();

  const bf16_t* b1base = L1 + (size_t)n0 * N_DIM;
  const bf16_t* b2base = L2 + (size_t)n0 * N_DIM;

  v8f acc1 = {};
  v8f acc2 = {};
#pragma unroll 2
  for (int k0 = 0; k0 < N_DIM; k0 += 32) {
    v16bf a  = load_a_frag((const bf16_t*)stem, N_DIM, lane, k0);
    v16bf f1 = load_b_frag(b1base,  N_DIM, lane, k0);
    v16bf f2 = load_b_frag(b2base,  N_DIM, lane, k0);
    acc1 = __builtin_amdgcn_wmma_f32_16x16x32_bf16(false, a, false, f1,
                                                   (short)0, acc1, false, false);
    acc2 = __builtin_amdgcn_wmma_f32_16x16x32_bf16(false, a, false, f2,
                                                   (short)0, acc2, false, false);
  }

  bf16_t* zb = Zt + (size_t)b * N_DIM * CC;
  {
    // D layout: lane holds column n = n0+(lane&15), rows c = c0+(lane>>4)*8+r
    const int n  = n0 + (lane & 15);
    const int cb = c0 + ((lane >> 4) << 3);
    bf16_t* zrow = zb + (size_t)n * CC;
#pragma unroll
    for (int r = 0; r < 8; ++r) {
      zrow[(cb + r) * 3 + 1] = (bf16_t)acc1[r];
      zrow[(cb + r) * 3 + 2] = (bf16_t)acc2[r];
    }
  }
  // k = 0: Lap[0] == I -> exact identity, transposed copy out of LDS
#pragma unroll
  for (int e = 0; e < 8; ++e) {
    const int lin = e * 32 + lane;   // 0..255 over the 16x16 tile
    const int cr  = lin >> 4;
    const int nn  = lin & 15;
    zb[(size_t)(n0 + nn) * CC + (c0 + cr) * 3] = stem[cr * N_DIM + n0 + nn];
  }
}

// ---- per-step kernel 2: gate GEMM + fused LSTM cell ---------------
// fea = W(256x288) @ Z(288x512) per batch; each wave computes the
// i/j/f/o tiles for one (ch-tile, n-tile) so the cell update fuses.
// All 8 waves of a block share ch0: the four 16x288 W row-blocks
// (36 KB) are staged in LDS once per block.
__global__ void __launch_bounds__(256) k_gates(
    const bf16_t* __restrict__ Zt, const bf16_t* __restrict__ Wbf,
    const float* __restrict__ bias, float* __restrict__ cstate,
    bf16_t* __restrict__ hbf, float* __restrict__ out, int t) {
  __shared__ __align__(16) bf16_t sw[4 * 16 * CC];    // 36 KB, [gate*16+row][288]

  const int lane = threadIdx.x & 31;
  const int wave = threadIdx.x >> 5;
  const int bid  = blockIdx.x;        // B * 4 * 4
  const int b      = bid >> 4;
  const int chtile = (bid >> 2) & 3;
  const int ng     = bid & 3;
  const int n0  = (ng * 8 + wave) * 16;
  const int ch0 = chtile * 16;

  // stage gate row-blocks: per gate 16x288 = 4608 bf16 = 576 x 16B chunks
#pragma unroll
  for (int s = 0; s < 4; ++s) {
    const bf16_t* src = Wbf + (size_t)(s * C_OUT + ch0) * CC;
    bf16_t* dst = sw + s * 16 * CC;
    for (int chunk = (int)threadIdx.x; chunk < 576; chunk += 256) {
      *(v8bf*)(dst + chunk * 8) = *(const v8bf*)(src + chunk * 8);
    }
  }
  __syncthreads();

  const bf16_t* zbase = Zt + (size_t)b * N_DIM * CC + (size_t)n0 * CC;

  v8f ai = {}, aj = {}, af = {}, ao = {};
#pragma unroll 3
  for (int k0 = 0; k0 < CC; k0 += 32) {
    v16bf fb = load_b_frag(zbase, CC, lane, k0);   // shared across 4 WMMAs
    v16bf a0 = load_a_frag((const bf16_t*)(sw + 0 * 16 * CC), CC, lane, k0);
    v16bf a1 = load_a_frag((const bf16_t*)(sw + 1 * 16 * CC), CC, lane, k0);
    v16bf a2 = load_a_frag((const bf16_t*)(sw + 2 * 16 * CC), CC, lane, k0);
    v16bf a3 = load_a_frag((const bf16_t*)(sw + 3 * 16 * CC), CC, lane, k0);
    ai = __builtin_amdgcn_wmma_f32_16x16x32_bf16(false, a0, false, fb, (short)0, ai, false, false);
    aj = __builtin_amdgcn_wmma_f32_16x16x32_bf16(false, a1, false, fb, (short)0, aj, false, false);
    af = __builtin_amdgcn_wmma_f32_16x16x32_bf16(false, a2, false, fb, (short)0, af, false, false);
    ao = __builtin_amdgcn_wmma_f32_16x16x32_bf16(false, a3, false, fb, (short)0, ao, false, false);
  }

  // fused bias + nonlinearities + c/h recurrence + output store
  const int n   = n0 + (lane & 15);
  const int chb = ch0 + ((lane >> 4) << 3);
#pragma unroll
  for (int r = 0; r < 8; ++r) {
    const int ch = chb + r;
    const float iv = ai[r] + bias[0 * C_OUT + ch];
    const float jv = aj[r] + bias[1 * C_OUT + ch];
    const float fv = af[r] + bias[2 * C_OUT + ch];
    const float ov = ao[r] + bias[3 * C_OUT + ch];
    const size_t idx = ((size_t)b * C_OUT + ch) * N_DIM + n;
    const float cp = cstate[idx];
    const float cn = cp * sigf(fv) + sigf(iv) * tanhf(jv);
    const float hn = tanhf(cn) * sigf(ov);
    cstate[idx] = cn;                    // fp32 recurrent cell state
    hbf[idx]    = (bf16_t)hn;            // bf16 h feeds next step's A rows
    out[idx * T_DIM + t] = hn;           // out[B,C_OUT,N,T] fp32
  }
}

// ------------------------------------------------------------------
extern "C" void kernel_launch(void* const* d_in, const int* in_sizes, int n_in,
                              void* d_out, int out_size, void* d_ws, size_t ws_size,
                              hipStream_t stream) {
  (void)in_sizes; (void)n_in; (void)out_size; (void)ws_size;
  const float* x    = (const float*)d_in[0];   // [64,32,512,24]
  const float* S    = (const float*)d_in[1];   // [512,512]
  const float* W    = (const float*)d_in[2];   // [256,288]
  const float* bias = (const float*)d_in[3];   // [256]
  float* out = (float*)d_out;                  // [64,64,512,24]

  char* p = (char*)d_ws;
  bf16_t* xbf = (bf16_t*)p; p += (size_t)T_DIM * B_DIM * C_IN * N_DIM * sizeof(bf16_t);
  bf16_t* Wbf = (bf16_t*)p; p += (size_t)4 * C_OUT * CC * sizeof(bf16_t);
  bf16_t* L1  = (bf16_t*)p; p += (size_t)N_DIM * N_DIM * sizeof(bf16_t);
  bf16_t* L2  = (bf16_t*)p; p += (size_t)N_DIM * N_DIM * sizeof(bf16_t);
  bf16_t* hbf = (bf16_t*)p; p += (size_t)B_DIM * C_OUT * N_DIM * sizeof(bf16_t);
  float*  cst = (float*)p;  p += (size_t)B_DIM * C_OUT * N_DIM * sizeof(float);
  bf16_t* Zt  = (bf16_t*)p; p += (size_t)B_DIM * N_DIM * CC * sizeof(bf16_t);

  // one-time prep (re-run every call: d_ws is poisoned, launch is deterministic)
  k_pack_x<<<(T_DIM * B_DIM * C_IN * N_DIM) / 256, 256, 0, stream>>>(x, xbf);
  k_pack_w<<<(4 * C_OUT * CC) / 256, 256, 0, stream>>>(W, Wbf);
  k_laps<<<(N_DIM * N_DIM) / 256, 256, 0, stream>>>(S, L1, L2);
  k_init_state<<<(B_DIM * C_OUT * N_DIM) / 256, 256, 0, stream>>>(hbf, cst);

  // serial recurrence over T; per-step working set stays L2-resident
  for (int t = 0; t < T_DIM; ++t) {
    k_spatial<<<B_DIM * 6 * 4, 256, 0, stream>>>(xbf, hbf, L1, L2, Zt, t);
    k_gates<<<B_DIM * 4 * 4, 256, 0, stream>>>(Zt, Wbf, bias, cst, hbf, out, t);
  }
}
